// TreeAttention_53541062312190
// MI455X (gfx1250) — compile-verified, matrix operand order checked by reading, not run on hardware
//
#include <hip/hip_runtime.h>
#include <math.h>

// ---------------------------------------------------------------------------
// TreeAttention on gfx1250 (MI455X).  Memory-bound gather/scatter segment
// softmax; the per-edge 3->8 MLP is mapped onto V_WMMA_F32_16X16X4_F32
// (16 edges per WMMA).  Segment reductions via float atomics (Q/K/V fit in
// the 192MB L2, so gathers are L2 hits; HBM traffic is streaming edge state).
// ---------------------------------------------------------------------------

#define TA_HEADS 8
#define TA_HD    8
#define TA_DIM   64
#define TA_SCALE 2.82842712474619f  // sqrt(8)
#define TA_BNEPS 1e-5f

typedef float v2f  __attribute__((ext_vector_type(2)));
typedef float v8f  __attribute__((ext_vector_type(8)));
typedef _Float16 v16h __attribute__((ext_vector_type(16)));

#if __has_builtin(__builtin_amdgcn_wmma_f32_16x16x4_f32)
#define TA_USE_WMMA_F32 1
#else
#define TA_USE_WMMA_F32 0
#endif

// ---- float atomic max via int punning (init value must be -inf) ------------
__device__ __forceinline__ void atomicMaxF(float* addr, float val) {
  if (val >= 0.0f)
    atomicMax((int*)addr, __float_as_int(val));
  else
    atomicMin((unsigned int*)addr, __float_as_uint(val));
}

// ---- K0: zero/seed all accumulators ----------------------------------------
__global__ void k_init(float* __restrict__ count, float* __restrict__ wmax,
                       float* __restrict__ denom, float* __restrict__ stats,
                       float* __restrict__ out, int N, int outN) {
  int i = blockIdx.x * blockDim.x + threadIdx.x;
  if (i < outN)  out[i]   = 0.0f;
  if (i < 8 * N) { wmax[i] = -__builtin_inff(); denom[i] = 0.0f; }
  if (i < N)     count[i] = 0.0f;
  if (i < 16)    stats[i] = 0.0f;
}

// ---- K1: per-edge h = (pq[q]-pk[k]) @ W1 + b1 ; BN partial sums ; counts ----
__global__ void k_edge_h(const float* __restrict__ pq, const float* __restrict__ pk,
                         const int* __restrict__ m, const float* __restrict__ W1,
                         const float* __restrict__ b1, float* __restrict__ hbuf,
                         float* __restrict__ count, float* __restrict__ stats, int E) {
  __shared__ float ls[6];
  if (threadIdx.x < 6) ls[threadIdx.x] = 0.0f;
  __syncthreads();

  int e = blockIdx.x * blockDim.x + threadIdx.x;
  float h0 = 0.0f, h1 = 0.0f, h2 = 0.0f;
  if (e < E) {
    int qi = m[2 * e], ki = m[2 * e + 1];
    float p0 = pq[3 * qi + 0] - pk[3 * ki + 0];
    float p1 = pq[3 * qi + 1] - pk[3 * ki + 1];
    float p2 = pq[3 * qi + 2] - pk[3 * ki + 2];
    h0 = fmaf(p0, W1[0], fmaf(p1, W1[3], fmaf(p2, W1[6], b1[0])));
    h1 = fmaf(p0, W1[1], fmaf(p1, W1[4], fmaf(p2, W1[7], b1[1])));
    h2 = fmaf(p0, W1[2], fmaf(p1, W1[5], fmaf(p2, W1[8], b1[2])));
    hbuf[3 * e + 0] = h0;
    hbuf[3 * e + 1] = h1;
    hbuf[3 * e + 2] = h2;
    atomicAdd(&count[qi], 1.0f);
  }
  // LDS float atomics (ds_add_f32) for the BN statistics block reduction.
  atomicAdd(&ls[0], h0);       atomicAdd(&ls[1], h1);       atomicAdd(&ls[2], h2);
  atomicAdd(&ls[3], h0 * h0);  atomicAdd(&ls[4], h1 * h1);  atomicAdd(&ls[5], h2 * h2);
  __syncthreads();
  if (threadIdx.x < 6) atomicAdd(&stats[threadIdx.x], ls[threadIdx.x]);
}

// ---- K2: finalize BN: scale/shift per channel -------------------------------
__global__ void k_bn(const float* __restrict__ gamma, const float* __restrict__ beta,
                     float* __restrict__ stats, float invE) {
  int c = threadIdx.x;
  if (c < 3) {
    float mu  = stats[c] * invE;
    float var = stats[3 + c] * invE - mu * mu;
    float sc  = gamma[c] * rsqrtf(var + TA_BNEPS);
    stats[8 + c]  = sc;                 // scale
    stats[11 + c] = beta[c] - mu * sc;  // shift
  }
}

// ---- K3: pe8 = relu(BN(h)) @ W2 + b2 via WMMA, 16 edges per instruction -----
__global__ void k_pe_wmma(const float* __restrict__ hbuf, const float* __restrict__ stats,
                          const float* __restrict__ W2, const float* __restrict__ b2,
                          float* __restrict__ pe8, int ntiles) {
  const int lane = threadIdx.x & 31;
  const int half = lane >> 4;   // 0: lanes 0-15, 1: lanes 16-31
  const int r    = lane & 15;
  const int wave   = (blockIdx.x * blockDim.x + threadIdx.x) >> 5;
  const int nwaves = (gridDim.x * blockDim.x) >> 5;

  const float sc0 = stats[8],  sh0 = stats[11];
  const float sc1 = stats[9],  sh1 = stats[12];
  const float sc2 = stats[10], sh2 = stats[13];
  const float bias = (r < 8) ? b2[r] : 0.0f;

#if TA_USE_WMMA_F32
  // B (4x16 f32): lane col n=r; v0 = row K=2*half, v1 = row K=2*half+1 (row 3 = pad 0).
  v2f B;
  B.x = (r < 8)              ? W2[(2 * half) * 8 + r]     : 0.0f;   // K=0 or K=2
  B.y = (r < 8 && half == 0) ? W2[1 * 8 + r]              : 0.0f;   // K=1 (K=3 is pad)
#else
  // B (32x16 f16): lanes 0-15 elems 0..2 = K0..2 (same slots as A), all else 0.
  v16h B;
  #pragma unroll
  for (int i = 0; i < 16; ++i) B[i] = (_Float16)0.0f;
  if (half == 0 && r < 8) {
    B[0] = (_Float16)W2[0 * 8 + r];
    B[1] = (_Float16)W2[1 * 8 + r];
    B[2] = (_Float16)W2[2 * 8 + r];
  }
#endif

  for (int t = wave; t < ntiles; t += nwaves) {
    const int e0 = t << 4;
    const int e  = e0 + r;
#if TA_USE_WMMA_F32
    // A (16x4 f32): row M=r; v0 = K=2*half, v1 = K=2*half+1 (K=3 pad 0).
    v2f A;
    if (half == 0) {
      A.x = fmaxf(fmaf(hbuf[3 * e + 0], sc0, sh0), 0.0f);
      A.y = fmaxf(fmaf(hbuf[3 * e + 1], sc1, sh1), 0.0f);
    } else {
      A.x = fmaxf(fmaf(hbuf[3 * e + 2], sc2, sh2), 0.0f);
      A.y = 0.0f;
    }
    v8f C = {0.0f, 0.0f, 0.0f, 0.0f, 0.0f, 0.0f, 0.0f, 0.0f};
    C = __builtin_amdgcn_wmma_f32_16x16x4_f32(false, A, false, B,
                                              (short)0, C, false, false);
#else
    // A (16x32 f16): lanes 0-15 elems 0..7 = K0..7 -> only K0..2 used.
    v16h A;
    #pragma unroll
    for (int i = 0; i < 16; ++i) A[i] = (_Float16)0.0f;
    if (half == 0) {
      A[0] = (_Float16)fmaxf(fmaf(hbuf[3 * e + 0], sc0, sh0), 0.0f);
      A[1] = (_Float16)fmaxf(fmaf(hbuf[3 * e + 1], sc1, sh1), 0.0f);
      A[2] = (_Float16)fmaxf(fmaf(hbuf[3 * e + 2], sc2, sh2), 0.0f);
    }
    v8f C = {0.0f, 0.0f, 0.0f, 0.0f, 0.0f, 0.0f, 0.0f, 0.0f};
    C = __builtin_amdgcn_wmma_f32_16x16x32_f16(false, A, false, B,
                                               (short)0, C, false, false);
#endif
    // D layout: VGPR v, lanes 0-15: M=v ; lanes 16-31: M=8+v ; N=lane&15.
    if (r < 8) {
      const int mb = e0 + half * 8;
      #pragma unroll
      for (int v = 0; v < 8; ++v)
        pe8[(mb + v) * 8 + r] = C[v] + bias;
    }
  }
}

// ---- K3b: scalar tail for E % 16 (not hit for E=1.6M, kept for safety) ------
__global__ void k_pe_scalar(const float* __restrict__ hbuf, const float* __restrict__ stats,
                            const float* __restrict__ W2, const float* __restrict__ b2,
                            float* __restrict__ pe8, int estart, int E) {
  int e = estart + blockIdx.x * blockDim.x + threadIdx.x;
  if (e >= E) return;
  float hn[3];
  #pragma unroll
  for (int c = 0; c < 3; ++c)
    hn[c] = fmaxf(fmaf(hbuf[3 * e + c], stats[8 + c], stats[11 + c]), 0.0f);
  #pragma unroll
  for (int d = 0; d < 8; ++d)
    pe8[e * 8 + d] = fmaf(hn[0], W2[d], fmaf(hn[1], W2[8 + d], fmaf(hn[2], W2[16 + d], b2[d])));
}

// ---- K4: per (edge,head) logit + segment max --------------------------------
// w[e,h] = (dot(q_h,k_h) + sum_d pe8[e,d]) / (SCALE*cnt[q] + 1e-8)
__global__ void k_logits(const float* __restrict__ Q, const float* __restrict__ K,
                         const int* __restrict__ m, const float* __restrict__ pe8,
                         const float* __restrict__ count, float* __restrict__ wbuf,
                         float* __restrict__ wmax, int EH) {
  int i = blockIdx.x * blockDim.x + threadIdx.x;
  if (i >= EH) return;
  const int e  = i >> 3;
  const int hh = i & 7;
  const int qi = m[2 * e], ki = m[2 * e + 1];
  const float* qrow = Q + (size_t)qi * TA_DIM + hh * TA_HD;
  const float* krow = K + (size_t)ki * TA_DIM + hh * TA_HD;
  float dot = 0.0f;
  #pragma unroll
  for (int d = 0; d < TA_HD; ++d) dot = fmaf(qrow[d], krow[d], dot);
  float ps = 0.0f;
  #pragma unroll
  for (int d = 0; d < TA_HD; ++d) ps += pe8[e * 8 + d];
  const float inv = 1.0f / (TA_SCALE * count[qi] + 1e-8f);
  const float w = (dot + ps) * inv;
  wbuf[i] = w;
  atomicMaxF(&wmax[qi * 8 + hh], w);
}

// ---- K5: exp(w - max) + segment sum -----------------------------------------
__global__ void k_expsum(const int* __restrict__ m, const float* __restrict__ wmax,
                         float* __restrict__ wbuf, float* __restrict__ denom, int EH) {
  int i = blockIdx.x * blockDim.x + threadIdx.x;
  if (i >= EH) return;
  const int e  = i >> 3;
  const int hh = i & 7;
  const int qi = m[2 * e];
  const float ex = expf(wbuf[i] - wmax[qi * 8 + hh]);
  wbuf[i] = ex;
  atomicAdd(&denom[qi * 8 + hh], ex);
}

// ---- K6: out[q] += (V[k] + pe8) * softmax -----------------------------------
__global__ void k_scatter_out(const float* __restrict__ Vv, const int* __restrict__ m,
                              const float* __restrict__ pe8, const float* __restrict__ wbuf,
                              const float* __restrict__ denom, float* __restrict__ out,
                              int EH) {
  int i = blockIdx.x * blockDim.x + threadIdx.x;
  if (i >= EH) return;
  const int e  = i >> 3;
  const int hh = i & 7;
  const int qi = m[2 * e], ki = m[2 * e + 1];
  const float a = wbuf[i] / denom[qi * 8 + hh];
  const float* vrow = Vv + (size_t)ki * TA_DIM + hh * TA_HD;
  __builtin_prefetch(vrow, 0, 1);  // global_prefetch_b8 warm-up of the V gather
  float* orow = out + (size_t)qi * TA_DIM + hh * TA_HD;
  #pragma unroll
  for (int d = 0; d < TA_HD; ++d)
    atomicAdd(&orow[d], (vrow[d] + pe8[e * 8 + d]) * a);
}

// ---------------------------------------------------------------------------
extern "C" void kernel_launch(void* const* d_in, const int* in_sizes, int n_in,
                              void* d_out, int out_size, void* d_ws, size_t ws_size,
                              hipStream_t stream) {
  const float* pq    = (const float*)d_in[0];
  const float* pk    = (const float*)d_in[1];
  const float* Q     = (const float*)d_in[2];
  const float* K     = (const float*)d_in[3];
  const float* V     = (const float*)d_in[4];
  const int*   m     = (const int*)d_in[5];
  const float* W1    = (const float*)d_in[6];
  const float* b1    = (const float*)d_in[7];
  const float* gamma = (const float*)d_in[8];
  const float* beta  = (const float*)d_in[9];
  const float* W2    = (const float*)d_in[10];
  const float* b2    = (const float*)d_in[11];
  float* out = (float*)d_out;

  const int E = in_sizes[5] / 2;
  const int N = in_sizes[2] / TA_DIM;

  // Workspace layout (floats)
  float* ws    = (float*)d_ws;
  float* stats = ws;                          // 16
  float* count = stats + 16;                  // N
  float* wmax  = count + N;                   // 8N
  float* denom = wmax + (size_t)8 * N;        // 8N
  float* hbuf  = denom + (size_t)8 * N;       // 3E
  float* pe8   = hbuf + (size_t)3 * E;        // 8E
  float* wbuf  = pe8 + (size_t)8 * E;         // 8E

  const int T = 256;
  const int EH = E * TA_HEADS;

  // K0: init
  const int initN = N * TA_DIM;
  k_init<<<(initN + T - 1) / T, T, 0, stream>>>(count, wmax, denom, stats, out, N, initN);

  // K1: h + BN stats + counts
  k_edge_h<<<(E + T - 1) / T, T, 0, stream>>>(pq, pk, m, W1, b1, hbuf, count, stats, E);

  // K2: BN finalize
  k_bn<<<1, 32, 0, stream>>>(gamma, beta, stats, 1.0f / (float)E);

  // K3: WMMA MLP (16 edges per tile)
  const int ntiles = E / 16;
  if (ntiles > 0) {
    int blocks = (ntiles + 7) / 8;  // 8 waves/block, 1 tile/wave pass
    k_pe_wmma<<<blocks, T, 0, stream>>>(hbuf, stats, W2, b2, pe8, ntiles);
  }
  const int rem = E - ntiles * 16;
  if (rem > 0)
    k_pe_scalar<<<(rem + T - 1) / T, T, 0, stream>>>(hbuf, stats, W2, b2, pe8, ntiles * 16, E);

  // K4: logits + segment max
  k_logits<<<(EH + T - 1) / T, T, 0, stream>>>(Q, K, m, pe8, count, wbuf, wmax, EH);

  // K5: exp + segment sum
  k_expsum<<<(EH + T - 1) / T, T, 0, stream>>>(m, wmax, wbuf, denom, EH);

  // K6: weighted scatter into output
  k_scatter_out<<<(EH + T - 1) / T, T, 0, stream>>>(V, m, pe8, wbuf, denom, out, EH);
}
